// Attention_48481590837653
// MI455X (gfx1250) — matrix operand order repaired and checked by reading, not compiled, not run on hardware
//
#include <hip/hip_runtime.h>
#include <hip/hip_bf16.h>
#include <math.h>

typedef __attribute__((ext_vector_type(16))) __bf16 v16bf;
typedef __attribute__((ext_vector_type(4)))  __bf16 v4bf;
typedef __attribute__((ext_vector_type(8)))  float  v8f;

#define NH   32      // query heads
#define NKV  8       // kv heads
#define REP  4       // NH / NKV
#define HD   128     // head dim
#define QT   16      // query rows per wave tile
#define KT   64      // keys per inner block (four 16-wide halves)

__device__ __forceinline__ v8f wmma_bf16(v16bf a, v16bf b, v8f c) {
  // D = A(16x32 bf16) * B(32x16 bf16) + C(16x16 f32)
  return __builtin_amdgcn_wmma_f32_16x16x32_bf16(false, a, false, b, (short)0, c,
                                                 false, false);
}

// ---------------- pre-pass 1: K fp32 -> bf16 (same layout) ----------------
__global__ void cvt_k_bf16(const float4* __restrict__ in, v4bf* __restrict__ o,
                           int n4) {
  const int i = blockIdx.x * blockDim.x + threadIdx.x;
  if (i < n4) {
    float4 f = in[i];
    v4bf b;
    b[0] = (__bf16)f.x; b[1] = (__bf16)f.y;
    b[2] = (__bf16)f.z; b[3] = (__bf16)f.w;
    o[i] = b;
  }
}

// ------- pre-pass 2: V fp32 [tok][kvh][d] -> bf16 vT [kvh*HD+d][tok] -------
__global__ void transpose_v_bf16(const float* __restrict__ v,
                                 __bf16* __restrict__ vt, int total) {
  const int tok = blockIdx.x * blockDim.x + threadIdx.x;
  const int row = blockIdx.y;            // kvh*HD + d
  if (tok < total) {
    const int kvh = row >> 7;            // / HD
    const int d   = row & (HD - 1);
    vt[(size_t)row * total + tok] =
        (__bf16)v[((size_t)tok * NKV + kvh) * HD + d];
  }
}

// ------------------------------ main kernel -------------------------------
__global__ __launch_bounds__(128, 1) void fa_varlen_fwd(
    const float* __restrict__ q, const __bf16* __restrict__ kb,
    const __bf16* __restrict__ vt, const int* __restrict__ cu,
    int n_seq, int total, float* __restrict__ out)
{
  __shared__ __bf16 lds_p[4][QT][KT];    // per-wave P transpose staging (8 KB)

  const int lane  = threadIdx.x & 31;
  const int wave  = threadIdx.x >> 5;
  const int lh    = lane & 15;
  const int hi    = lane >> 4;
  const int abase = hi * 8;

  const int kvh = blockIdx.y;
  const int qh  = kvh * REP + wave;

  const int tok0 = blockIdx.x * QT;
  if (tok0 >= total) return;

  int b = 0;
  for (int i = 0; i < n_seq - 1; ++i)
    if (tok0 >= cu[i + 1]) b = i + 1;
  const int s0   = cu[b];
  const int lend = cu[b + 1];
  const int len  = lend - s0;
  const int p0   = tok0 - s0;

  // ---- load Q tile (16 x 128 fp32 -> bf16) as 4 A-fragments (once) ----
  const int qtok = min(tok0 + lh, total - 1);
  const float* qrow = q + ((size_t)qtok * NH + qh) * HD;
  v16bf qa[4];
  #pragma unroll
  for (int c = 0; c < 4; ++c) {
    #pragma unroll
    for (int p = 0; p < 8; ++p) {
      const int kk = p * 2 + abase * 2;
      float2 f = *(const float2*)(qrow + c * 32 + kk);
      qa[c][2 * p]     = (__bf16)f.x;
      qa[c][2 * p + 1] = (__bf16)f.y;
    }
  }

  v16bf ones;
  #pragma unroll
  for (int i = 0; i < 16; ++i) ones[i] = (__bf16)1.0f;

  v8f oacc[8];
  #pragma unroll
  for (int c = 0; c < 8; ++c) oacc[c] = {};
  float mrow[8], lrow[8];
  #pragma unroll
  for (int r = 0; r < 8; ++r) { mrow[r] = -INFINITY; lrow[r] = 0.f; }

  const float sc = 0.08838834764831845f * 1.44269504088896340736f;

  const int kblocks = (p0 + QT + KT - 1) / KT;
  // blocks with n0+KT-1 <= p0 need no causal/len masking
  int nfull = (p0 >= KT - 1) ? ((p0 - (KT - 1)) / KT + 1) : 0;
  if (nfull > kblocks) nfull = kblocks;

  for (int j = 0; j < kblocks; ++j) {
    const int  n0     = j * KT;
    const bool masked = (j >= nfull);    // uniform across the wave

    // ---- S = Q @ K^T for 64 keys (four 16x16 fragments) ----
    v8f sfrag[4];
    #pragma unroll
    for (int h = 0; h < 4; ++h) sfrag[h] = {};
    #pragma unroll
    for (int half = 0; half < 4; ++half) {
      const int kpos = min(n0 + half * 16 + lh, len - 1);   // clamp (masked later)
      const __bf16* krow = kb + ((size_t)(s0 + kpos) * NKV + kvh) * HD;
      #pragma unroll
      for (int c = 0; c < 4; ++c) {
        v16bf kf = *(const v16bf*)(krow + c * 32 + hi * 16); // 32B contiguous
        sfrag[half] = wmma_bf16(qa[c], kf, sfrag[half]);
      }
    }

    // ---- scale (+ mask only on diagonal/tail blocks) ----
    float xs[4][8];
    #pragma unroll
    for (int h = 0; h < 4; ++h)
      #pragma unroll
      for (int r = 0; r < 8; ++r) xs[h][r] = sfrag[h][r] * sc;
    if (masked) {
      #pragma unroll
      for (int h = 0; h < 4; ++h) {
        const int kx = n0 + h * 16 + lh;
        #pragma unroll
        for (int r = 0; r < 8; ++r) {
          const int qpos = p0 + r + hi * 8;
          if (kx > qpos || kx >= len) xs[h][r] = -INFINITY;
        }
      }
    }

    // ---- online max (DS-permute tree, 8 rows x 4 levels) ----
    float mnew[8], corr[8];
    #pragma unroll
    for (int r = 0; r < 8; ++r) {
      float mx = fmaxf(fmaxf(xs[0][r], xs[1][r]), fmaxf(xs[2][r], xs[3][r]));
      mx = fmaxf(mx, __shfl_xor(mx, 1, 32));
      mx = fmaxf(mx, __shfl_xor(mx, 2, 32));
      mx = fmaxf(mx, __shfl_xor(mx, 4, 32));
      mx = fmaxf(mx, __shfl_xor(mx, 8, 32));
      mnew[r] = fmaxf(mrow[r], mx);
      corr[r] = exp2f(mrow[r] - mnew[r]);
      mrow[r] = mnew[r];
    }

    // ---- P = exp2(x - m), staged to LDS (C-layout -> A-layout) ----
    #pragma unroll
    for (int h = 0; h < 4; ++h)
      #pragma unroll
      for (int r = 0; r < 8; ++r)
        lds_p[wave][r + hi * 8][h * 16 + lh] = (__bf16)exp2f(xs[h][r] - mnew[r]);

    // rescale running O while P stores are in flight
    #pragma unroll
    for (int c = 0; c < 8; ++c)
      #pragma unroll
      for (int r = 0; r < 8; ++r) oacc[c][r] *= corr[r];

    asm volatile("s_wait_dscnt 0" ::: "memory");
    v16bf pa0, pa1;
    #pragma unroll
    for (int p = 0; p < 8; ++p) {
      const int kk = p * 2 + abase * 2;
      union { unsigned int u; __bf16 h[2]; } c0, c1;
      c0.u = *(const unsigned int*)&lds_p[wave][lh][kk];
      c1.u = *(const unsigned int*)&lds_p[wave][lh][kk + 32];
      pa0[2 * p] = c0.h[0]; pa0[2 * p + 1] = c0.h[1];
      pa1[2 * p] = c1.h[0]; pa1[2 * p + 1] = c1.h[1];
    }

    // ---- row sums via WMMA: l += P @ ones ----
    v8f lz = {};
    lz = wmma_bf16(pa0, ones, lz);
    lz = wmma_bf16(pa1, ones, lz);
    #pragma unroll
    for (int r = 0; r < 8; ++r) lrow[r] = lrow[r] * corr[r] + lz[r];

    // ---- O += P @ V (bf16 vT: contiguous 32B loads along tokens) ----
    const int t0 = s0 + min(n0 + hi * 16,      max(0, len - 16));
    const int t1 = s0 + min(n0 + 32 + hi * 16, max(0, len - 16));
    #pragma unroll
    for (int c = 0; c < 8; ++c) {
      const size_t vrow = (size_t)(kvh * HD + c * 16 + lh) * total;
      v16bf vb0 = *(const v16bf*)(vt + vrow + t0);
      v16bf vb1 = *(const v16bf*)(vt + vrow + t1);
      oacc[c] = wmma_bf16(pa0, vb0, oacc[c]);
      oacc[c] = wmma_bf16(pa1, vb1, oacc[c]);
    }
  }

  // ---- epilogue: O / l, scatter back to varlen layout ----
  float rinv[8];
  #pragma unroll
  for (int r = 0; r < 8; ++r) rinv[r] = 1.0f / lrow[r];
  #pragma unroll
  for (int c = 0; c < 8; ++c) {
    const int d = c * 16 + lh;
    #pragma unroll
    for (int r = 0; r < 8; ++r) {
      const int tok = tok0 + r + hi * 8;
      if (tok < lend)
        out[((size_t)tok * NH + qh) * HD + d] = oacc[c][r] * rinv[r];
    }
  }
}

extern "C" void kernel_launch(void* const* d_in, const int* in_sizes, int n_in,
                              void* d_out, int out_size, void* d_ws, size_t ws_size,
                              hipStream_t stream) {
  const float* q  = (const float*)d_in[0];
  const float* k  = (const float*)d_in[1];
  const float* v  = (const float*)d_in[2];
  const int*   cu = (const int*)d_in[3];
  const int n_seq = in_sizes[3] - 1;
  const int total = in_sizes[0] / (NH * HD);

  const int nkv = total * NKV * HD;          // elements in K (and V)
  __bf16* kb = (__bf16*)d_ws;                // bf16 K, same layout
  __bf16* vt = kb + (size_t)nkv;             // bf16 V, transposed [kvh*HD+d][tok]

  // pre-pass: one-time bf16 conversion (+ V transpose)
  cvt_k_bf16<<<dim3((nkv / 4 + 255) / 256), dim3(256), 0, stream>>>(
      (const float4*)k, (v4bf*)kb, nkv / 4);
  transpose_v_bf16<<<dim3((total + 255) / 256, NKV * HD), dim3(256), 0, stream>>>(
      v, vt, total);

  dim3 grid((total + QT - 1) / QT, NKV);
  fa_varlen_fwd<<<grid, dim3(128), 0, stream>>>(q, kb, vt, cu, n_seq, total,
                                                (float*)d_out);
}
